// CausalSelfAttention_2654289789124
// MI455X (gfx1250) — matrix operand order repaired and checked by reading, not compiled, not run on hardware
//
#include <hip/hip_runtime.h>
#include <hip/hip_bf16.h>

#define NH   16
#define NKV  4
#define HD   128
#define SEQ  2048
#define CDIM 2048
#define BATCH 2

typedef __attribute__((ext_vector_type(16))) __bf16 bf16x16;
typedef __attribute__((ext_vector_type(8)))  float  f32x8;
typedef __attribute__((ext_vector_type(4)))  unsigned int u32x4;
typedef __attribute__((ext_vector_type(8)))  unsigned int u32x8;
typedef unsigned short u16;

// ---------- helpers ----------

__device__ __forceinline__ u16 f2bf(float f) {
    unsigned int u = __float_as_uint(f);
    u += 0x7FFFu + ((u >> 16) & 1u);   // round-to-nearest-even
    return (u16)(u >> 16);
}

__device__ __forceinline__ bf16x16 make_frag(u32x4 lo, u32x4 hi) {
    u32x8 t;
    t[0] = lo[0]; t[1] = lo[1]; t[2] = lo[2]; t[3] = lo[3];
    t[4] = hi[0]; t[5] = hi[1]; t[6] = hi[2]; t[7] = hi[3];
    return __builtin_bit_cast(bf16x16, t);
}

// A fragment (16xK tile, K=32): lane holds row = lane&15,
// halves 0..7 -> K = kb..kb+7, halves 8..15 -> K = kb+16..kb+23, kb = (lane>>4)*8
__device__ __forceinline__ bf16x16 load_afrag(const u16* base, int stride, int lane) {
    const int row = lane & 15;
    const int kb  = (lane >> 4) << 3;
    const u16* p = base + row * stride + kb;
    u32x4 lo = *(const u32x4*)(p);
    u32x4 hi = *(const u32x4*)(p + 16);
    return make_frag(lo, hi);
}

// B fragment (Kx16 tile, K=32): lane holds col = lane&15,
// halves 0..7 -> K = kb..kb+7, halves 8..15 -> K = kb+16..kb+23, kb = (lane>>4)*8
__device__ __forceinline__ bf16x16 load_bfrag(const u16* base, int stride, int lane) {
    const int col = lane & 15;
    const int kb  = (lane >> 4) << 3;
    const u16* p = base + col * stride + kb;
    u32x4 lo = *(const u32x4*)(p);
    u32x4 hi = *(const u32x4*)(p + 16);
    return make_frag(lo, hi);
}

__device__ __forceinline__ f32x8 wmma_bf16(bf16x16 a, bf16x16 b, f32x8 c) {
    return __builtin_amdgcn_wmma_f32_16x16x32_bf16(false, a, false, b, (short)0, c, false, false);
}

// CDNA5 async copy: global -> LDS, 16 bytes per lane, tracked with ASYNCcnt.
// LDS operand = wave-relative LDS offset (low 32 bits of generic shared ptr).
__device__ __forceinline__ void async_copy_b128(void* lds_ptr, const void* gptr) {
    unsigned lds_off = (unsigned)(uintptr_t)lds_ptr;
    asm volatile("global_load_async_to_lds_b128 %0, %1, off"
                 :: "v"(lds_off), "v"(gptr)
                 : "memory");
}

// ---------- elementwise kernels ----------

__global__ __launch_bounds__(256) void convert_f32_bf16(const float* __restrict__ in,
                                                        u16* __restrict__ out, int n) {
    int i = blockIdx.x * 256 + threadIdx.x;
    if (i < n) out[i] = f2bf(in[i]);
}

// RoPE (non-adjacent-pairs): out[d] = x[d]*cos + rot(x)[d]*sin
__global__ __launch_bounds__(256) void rope_f32_bf16(const float* __restrict__ in,
                                                     u16* __restrict__ out,
                                                     int nheads, int n) {
    int i = blockIdx.x * 256 + threadIdx.x;
    if (i >= n) return;
    int d   = i & (HD - 1);
    int row = i >> 7;                      // (b*SEQ + t)*nheads + h
    int t   = (row / nheads) & (SEQ - 1);
    int fd  = d & 63;
    // inv_freq = 10000^(-fd/64) = exp(-fd * ln(10000)/64)
    float inv_freq = __expf(-(float)fd * 0.14391156831212788f);
    float ang = (float)t * inv_freq;
    float s, c;
    __sincosf(ang, &s, &c);
    float x   = in[i];
    float rot = (d < 64) ? -in[i + 64] : in[i - 64];
    out[i] = f2bf(x * c + rot * s);
}

// ---------- GEMM: C[M,N] f32 = A[M,K] @ B[N,K]^T, A/B bf16 row-major ----------
// Block tile 256x128, 8 waves (4M x 2N), wave tile 64x64 (4x4 WMMA accumulators).
// Double-buffered LDS fed by global_load_async_to_lds_b128 (ASYNCcnt pipeline):
// 16 WMMAs per 16 fragment ds_loads per barrier pair.

#define LDA 40   // padded LDS stride (halves), 16B-aligned, breaks bank conflicts

__global__ __launch_bounds__(256) void gemm_bf16_nt(const u16* __restrict__ A,
                                                    const u16* __restrict__ Bw,
                                                    float* __restrict__ C,
                                                    int M, int N, int K) {
    __shared__ __align__(16) u16 As[2][256 * LDA];   // 2 x 20 KB
    __shared__ __align__(16) u16 Bs[2][128 * LDA];   // 2 x 10 KB

    const int tid  = threadIdx.x;
    const int lane = tid & 31;
    const int wave = tid >> 5;
    const int bm   = blockIdx.y * 256;
    const int bn   = blockIdx.x * 128;
    const int wm   = (wave >> 1) * 64;   // 0/64/128/192
    const int wn   = (wave & 1) * 64;    // 0/64

    const f32x8 zero = {0.f, 0.f, 0.f, 0.f, 0.f, 0.f, 0.f, 0.f};
    f32x8 acc[4][4];
#pragma unroll
    for (int mi = 0; mi < 4; ++mi)
#pragma unroll
        for (int ni = 0; ni < 4; ++ni) acc[mi][ni] = zero;

    // staging: A = 256 rows x 32 halves (1 row / thread, 4 async b128),
    //          B = 128 rows x 32 halves (half row / thread, 2 async b128)
    const int arow = tid;                 // 0..255
    const int brow = tid >> 1;            // 0..127
    const int bseg = (tid & 1) * 16;      // halves
    const u16* pa = A  + (size_t)(bm + arow) * K;
    const u16* pb = Bw + (size_t)(bn + brow) * K + bseg;
    const int aoff = arow * LDA;
    const int boff = brow * LDA + bseg;

    // prologue: tile 0 in flight
#pragma unroll
    for (int s = 0; s < 4; ++s) async_copy_b128(&As[0][aoff + s * 8], pa + s * 8);
    async_copy_b128(&Bs[0][boff],     pb);
    async_copy_b128(&Bs[0][boff + 8], pb + 8);

    const int niter = K >> 5;
    for (int it = 0; it < niter; ++it) {
        const int buf = it & 1;
        if (it + 1 < niter) {
            const int nb = buf ^ 1;
            const u16* pa2 = pa + (size_t)(it + 1) * 32;
            const u16* pb2 = pb + (size_t)(it + 1) * 32;
#pragma unroll
            for (int s = 0; s < 4; ++s) async_copy_b128(&As[nb][aoff + s * 8], pa2 + s * 8);
            async_copy_b128(&Bs[nb][boff],     pb2);
            async_copy_b128(&Bs[nb][boff + 8], pb2 + 8);
            // in-order completion: <=6 outstanding means tile `it` has landed
            asm volatile("s_wait_asynccnt 0x6" ::: "memory");
        } else {
            asm volatile("s_wait_asynccnt 0x0" ::: "memory");
        }
        __syncthreads();

        bf16x16 af[4], bfr[4];
#pragma unroll
        for (int mi = 0; mi < 4; ++mi)
            af[mi] = load_afrag(&As[buf][(wm + mi * 16) * LDA], LDA, lane);
#pragma unroll
        for (int ni = 0; ni < 4; ++ni)
            bfr[ni] = load_bfrag(&Bs[buf][(wn + ni * 16) * LDA], LDA, lane);
#pragma unroll
        for (int mi = 0; mi < 4; ++mi)
#pragma unroll
            for (int ni = 0; ni < 4; ++ni)
                acc[mi][ni] = wmma_bf16(af[mi], bfr[ni], acc[mi][ni]);
        __syncthreads();   // readers done before next tile's async overwrites buf
    }

    const int rb = (lane >= 16) ? 8 : 0;
    const int cn = lane & 15;
#pragma unroll
    for (int mi = 0; mi < 4; ++mi)
#pragma unroll
        for (int ni = 0; ni < 4; ++ni)
#pragma unroll
            for (int r = 0; r < 8; ++r) {
                int rrow = bm + wm + mi * 16 + rb + r;
                int ccol = bn + wn + ni * 16 + cn;
                C[(size_t)rrow * N + ccol] = acc[mi][ni][r];
            }
}

// ---------- Flash attention (GQA, causal, online softmax) ----------
// grid: (SEQ/128, NH, BATCH), block 256 (8 waves, 16 q-rows each).
// Key tile = 64: 32 WMMAs per wave per barrier pair (16 for S, 16 for P.V).

#define LDK 136  // K-tile LDS stride (halves): 128 + pad, 16B aligned
#define LDV 72   // Vt / P LDS stride (halves): 64 + pad, 16B aligned

__global__ __launch_bounds__(256) void flash_attn(const u16* __restrict__ qb,
                                                  const u16* __restrict__ kb,
                                                  const u16* __restrict__ vb,
                                                  u16* __restrict__ yb) {
    __shared__ __align__(16) u16 Ks[64 * LDK];       // K tile [key][d]     (~17 KB)
    __shared__ __align__(16) u16 Vt[128 * LDV];      // V tile transposed [d][key] (~18 KB)
    __shared__ __align__(16) u16 Ps[8][16 * LDV];    // per-wave P tile [qrow][key] (~18 KB)

    const int tid  = threadIdx.x;
    const int lane = tid & 31;
    const int wave = tid >> 5;
    const int qblk = blockIdx.x * 128;
    const int h    = blockIdx.y;
    const int b    = blockIdx.z;
    const int kvh  = h >> 2;                         // GQA: 4 q-heads per kv-head

    // Q fragments for this wave's 16 rows (full head dim, 4 K-chunks of 32)
    const int qrow = qblk + wave * 16 + (lane & 15);
    const u16* qp  = qb + ((size_t)(b * SEQ + qrow) * NH + h) * HD + ((lane >> 4) << 3);
    bf16x16 aq[4];
#pragma unroll
    for (int kc = 0; kc < 4; ++kc) {
        u32x4 lo = *(const u32x4*)(qp + kc * 32);
        u32x4 hi = *(const u32x4*)(qp + kc * 32 + 16);
        aq[kc] = make_frag(lo, hi);
    }

    const f32x8 zero = {0.f, 0.f, 0.f, 0.f, 0.f, 0.f, 0.f, 0.f};
    f32x8 oacc[8];
#pragma unroll
    for (int i = 0; i < 8; ++i) oacc[i] = zero;
    float mrow[8], lrow[8];
#pragma unroll
    for (int r = 0; r < 8; ++r) { mrow[r] = -3.0e38f; lrow[r] = 0.f; }

    const int   rb      = (lane >= 16) ? 8 : 0;
    const int   qr_base = qblk + wave * 16 + rb;
    const int   cn      = lane & 15;
    const float scale   = 0.08838834764831845f;      // 1/sqrt(128)

    // per-thread staging coordinates
    const int kkey = tid >> 2, kseg = (tid & 3) * 32;     // K tile: 64 rows x 4 segs(32h)
    const int vkey = tid & 63, vd0  = (tid >> 6) * 32;    // V tile: 64 rows x 4 d-groups(32)

    const int nkt = (qblk >> 6) + 2;                 // causal: key tiles of 64
    for (int kt = 0; kt < nkt; ++kt) {
        // ---- stage K tile [64][128] via async global->LDS copies ----
        {
            const u16* p = kb + ((size_t)(b * SEQ + kt * 64 + kkey) * NKV + kvh) * HD + kseg;
#pragma unroll
            for (int s = 0; s < 4; ++s)
                async_copy_b128(&Ks[kkey * LDK + kseg + s * 8], p + s * 8);
        }
        // ---- stage V tile transposed [128][64] (VGPR path; needs transpose) ----
        {
            const u16* p = vb + ((size_t)(b * SEQ + kt * 64 + vkey) * NKV + kvh) * HD + vd0;
            if (kt + 1 < nkt) __builtin_prefetch(p + 64 * NKV * HD, 0, 3);
            union { u32x4 v[4]; u16 hh[32]; } tmp;
#pragma unroll
            for (int s = 0; s < 4; ++s) tmp.v[s] = *(const u32x4*)(p + s * 8);
#pragma unroll
            for (int i = 0; i < 32; ++i) Vt[(vd0 + i) * LDV + vkey] = tmp.hh[i];
        }
        asm volatile("s_wait_asynccnt 0x0" ::: "memory");
        __syncthreads();

        // ---- S = Q K^T : 16x64 per wave (4 N-tiles), K-dim = 128 ----
        f32x8 sacc[4];
#pragma unroll
        for (int nt = 0; nt < 4; ++nt) sacc[nt] = zero;
#pragma unroll
        for (int kc = 0; kc < 4; ++kc)
#pragma unroll
            for (int nt = 0; nt < 4; ++nt) {
                bf16x16 bk = load_bfrag(&Ks[nt * 16 * LDK + kc * 32], LDK, lane);
                sacc[nt] = wmma_bf16(aq[kc], bk, sacc[nt]);
            }

        // ---- online softmax (row stats live in each 16-lane half) ----
#pragma unroll
        for (int r = 0; r < 8; ++r) {
            const int qr = qr_base + r;
            float sv[4];
            bool  msk[4];
#pragma unroll
            for (int nt = 0; nt < 4; ++nt) {
                const int ki = kt * 64 + nt * 16 + cn;
                msk[nt] = ki > qr;
                sv[nt]  = msk[nt] ? -1.0e30f : sacc[nt][r] * scale;
            }
            float mx = fmaxf(fmaxf(sv[0], sv[1]), fmaxf(sv[2], sv[3]));
            mx = fmaxf(mx, __shfl_xor(mx, 1, 32));
            mx = fmaxf(mx, __shfl_xor(mx, 2, 32));
            mx = fmaxf(mx, __shfl_xor(mx, 4, 32));
            mx = fmaxf(mx, __shfl_xor(mx, 8, 32));
            const float mnew = fmaxf(mrow[r], mx);
            const float corr = __expf(mrow[r] - mnew);
            mrow[r] = mnew;
            float pv[4], rs = 0.f;
#pragma unroll
            for (int nt = 0; nt < 4; ++nt) {
                pv[nt] = msk[nt] ? 0.f : __expf(sv[nt] - mnew);
                rs += pv[nt];
            }
            rs += __shfl_xor(rs, 1, 32);
            rs += __shfl_xor(rs, 2, 32);
            rs += __shfl_xor(rs, 4, 32);
            rs += __shfl_xor(rs, 8, 32);
            lrow[r] = lrow[r] * corr + rs;
#pragma unroll
            for (int nt = 0; nt < 8; ++nt) oacc[nt][r] *= corr;
#pragma unroll
            for (int nt = 0; nt < 4; ++nt)
                Ps[wave][(rb + r) * LDV + nt * 16 + cn] = f2bf(pv[nt]);
        }

        // same-wave LDS store -> load: drain DS counter
        asm volatile("s_wait_dscnt 0x0" ::: "memory");

        // ---- O += P V : A = P (16x64, 2 K-chunks), B = Vt columns ----
        bf16x16 ap[2];
#pragma unroll
        for (int kc = 0; kc < 2; ++kc)
            ap[kc] = load_afrag(&Ps[wave][kc * 32], LDV, lane);
#pragma unroll
        for (int nt = 0; nt < 8; ++nt)
#pragma unroll
            for (int kc = 0; kc < 2; ++kc) {
                bf16x16 bv = load_bfrag(&Vt[nt * 16 * LDV + kc * 32], LDV, lane);
                oacc[nt] = wmma_bf16(ap[kc], bv, oacc[nt]);
            }
        __syncthreads();
    }

    // ---- epilogue: normalize and store y[b,t,h*128+d] as bf16 ----
#pragma unroll
    for (int r = 0; r < 8; ++r) {
        const float inv = 1.0f / lrow[r];
        const int t = qr_base + r;
        u16* yp = yb + (size_t)(b * SEQ + t) * CDIM + h * HD;
#pragma unroll
        for (int nt = 0; nt < 8; ++nt)
            yp[nt * 16 + cn] = f2bf(oacc[nt][r] * inv);
    }
}

// ---------- launcher ----------

extern "C" void kernel_launch(void* const* d_in, const int* in_sizes, int n_in,
                              void* d_out, int out_size, void* d_ws, size_t ws_size,
                              hipStream_t stream) {
    const float* x  = (const float*)d_in[0];
    const float* Wq = (const float*)d_in[1];
    const float* Wk = (const float*)d_in[2];
    const float* Wv = (const float*)d_in[3];
    const float* Wo = (const float*)d_in[4];
    float* out = (float*)d_out;

    char* ws = (char*)d_ws;
    size_t off = 0;
    auto alloc = [&](size_t bytes) -> void* {
        void* p = ws + off;
        off += (bytes + 255) & ~(size_t)255;
        return p;
    };

    const int Mrows = BATCH * SEQ;        // 4096
    const int KVC   = NKV * HD;           // 512

    u16*   x_bf  = (u16*)  alloc((size_t)Mrows * CDIM * 2);
    u16*   wq_bf = (u16*)  alloc((size_t)CDIM * CDIM * 2);
    u16*   wk_bf = (u16*)  alloc((size_t)KVC * CDIM * 2);
    u16*   wv_bf = (u16*)  alloc((size_t)KVC * CDIM * 2);
    u16*   wo_bf = (u16*)  alloc((size_t)CDIM * CDIM * 2);
    float* q_f   = (float*)alloc((size_t)Mrows * CDIM * 4);
    float* k_f   = (float*)alloc((size_t)Mrows * KVC * 4);
    float* v_f   = (float*)alloc((size_t)Mrows * KVC * 4);
    u16*   q_bf  = (u16*)  alloc((size_t)Mrows * CDIM * 2);
    u16*   k_bf  = (u16*)  alloc((size_t)Mrows * KVC * 2);
    u16*   v_bf  = (u16*)  alloc((size_t)Mrows * KVC * 2);
    u16*   y_bf  = (u16*)  q_f;  // alias: q_f32 is dead once RoPE has consumed it

    auto cdiv = [](int a, int b) { return (a + b - 1) / b; };

    // 1) f32 -> bf16 conversions
    int n;
    n = Mrows * CDIM; convert_f32_bf16<<<cdiv(n, 256), 256, 0, stream>>>(x,  x_bf,  n);
    n = CDIM * CDIM;  convert_f32_bf16<<<cdiv(n, 256), 256, 0, stream>>>(Wq, wq_bf, n);
    n = KVC * CDIM;   convert_f32_bf16<<<cdiv(n, 256), 256, 0, stream>>>(Wk, wk_bf, n);
    n = KVC * CDIM;   convert_f32_bf16<<<cdiv(n, 256), 256, 0, stream>>>(Wv, wv_bf, n);
    n = CDIM * CDIM;  convert_f32_bf16<<<cdiv(n, 256), 256, 0, stream>>>(Wo, wo_bf, n);

    // 2) projections (WMMA GEMMs, async-LDS double-buffered)
    gemm_bf16_nt<<<dim3(CDIM / 128, Mrows / 256), 256, 0, stream>>>(x_bf, wq_bf, q_f, Mrows, CDIM, CDIM);
    gemm_bf16_nt<<<dim3(KVC  / 128, Mrows / 256), 256, 0, stream>>>(x_bf, wk_bf, k_f, Mrows, KVC,  CDIM);
    gemm_bf16_nt<<<dim3(KVC  / 128, Mrows / 256), 256, 0, stream>>>(x_bf, wv_bf, v_f, Mrows, KVC,  CDIM);

    // 3) RoPE (q, k) + convert (v) to bf16
    n = Mrows * CDIM; rope_f32_bf16<<<cdiv(n, 256), 256, 0, stream>>>(q_f, q_bf, NH,  n);
    n = Mrows * KVC;  rope_f32_bf16<<<cdiv(n, 256), 256, 0, stream>>>(k_f, k_bf, NKV, n);
    n = Mrows * KVC;  convert_f32_bf16<<<cdiv(n, 256), 256, 0, stream>>>(v_f, v_bf, n);

    // 4) causal flash attention (WMMA), writes y[b,t,h,d] bf16
    flash_attn<<<dim3(SEQ / 128, NH, BATCH), 256, 0, stream>>>(q_bf, k_bf, v_bf, y_bf);

    // 5) output projection -> f32 out
    gemm_bf16_nt<<<dim3(CDIM / 128, Mrows / 256), 256, 0, stream>>>(y_bf, wo_bf, out, Mrows, CDIM, CDIM);
}
